// DecoderTrans_86870008528896
// MI455X (gfx1250) — matrix operand order repaired and checked by reading, not compiled, not run on hardware
//
#include <hip/hip_runtime.h>

#define DMODEL 512
#define NHEAD 8
#define DHEAD 64
#define TLEN 1024
#define SLEN 1024
#define BATCH 4
#define VOCAB 32000
#define NROWS (BATCH * TLEN)   // 4096

typedef __attribute__((ext_vector_type(16))) __bf16 v16bf;
typedef __attribute__((ext_vector_type(8)))  float  v8f;

union FragBF {
  v16bf v;
  uint4 q[2];
  unsigned short u[16];
};

__device__ inline unsigned short f2bf(float x) {
  unsigned int u = __float_as_uint(x);
  unsigned int r = u + 0x7fffu + ((u >> 16) & 1u);   // round-to-nearest-even
  return (unsigned short)(r >> 16);
}

__device__ inline v8f zero8() {
  v8f z;
  for (int r = 0; r < 8; ++r) z[r] = 0.0f;
  return z;
}

// A fragment (16x32, 16-bit): per lane two 8-element chunks at K offsets {8*hs, 16+8*hs}
__device__ inline FragBF ldfragA(const unsigned short* p) {
  FragBF f;
  f.q[0] = *(const uint4*)(p);
  f.q[1] = *(const uint4*)(p + 16);
  return f;
}
// B fragment (32x16, 16-bit): per lane two 8-element chunks at K offsets {8*hs, 16+8*hs}
__device__ inline FragBF ldfragB(const unsigned short* p) {
  FragBF f;
  f.q[0] = *(const uint4*)(p);
  f.q[1] = *(const uint4*)(p + 16);
  return f;
}

__device__ inline v8f wmma_bf16(const FragBF& a, const FragBF& b, v8f c) {
  return __builtin_amdgcn_wmma_f32_16x16x32_bf16(
      /*neg_a=*/false, a.v, /*neg_b=*/false, b.v,
      /*c_mod=*/(short)0, c, /*reuse_a=*/false, /*reuse_b=*/false);
}

// ---------------------------------------------------------------------------
// fp32 -> bf16 convert
// ---------------------------------------------------------------------------
__global__ void f2bf_kernel(const float* __restrict__ in,
                            unsigned short* __restrict__ out, size_t n) {
  size_t i = (size_t)blockIdx.x * blockDim.x + threadIdx.x;
  if (i < n) out[i] = f2bf(in[i]);
}

// ---------------------------------------------------------------------------
// Embedding + sinusoidal positional encoding. One wave per token row.
// ---------------------------------------------------------------------------
__global__ __launch_bounds__(256) void embed_kernel(
    const int* __restrict__ tokens, const float* __restrict__ emb,
    float* __restrict__ xf, unsigned short* __restrict__ xb) {
  int row = blockIdx.x * 8 + (threadIdx.x >> 5);
  int lane = threadIdx.x & 31;
  int t = row % TLEN;
  int tok = tokens[row];
  const float c = 9.210340371976184f / (float)DMODEL;  // ln(10000)/D
  size_t base = (size_t)row * DMODEL;
  for (int j = 0; j < DMODEL / 32; ++j) {
    int d = j * 32 + lane;
    float f = __expf(-(float)(d & ~1) * c);
    float ang = (float)t * f;
    float pe = (d & 1) ? cosf(ang) : sinf(ang);
    float v = emb[(size_t)tok * DMODEL + d] * 22.62741699796952f + pe;  // sqrt(512)
    xf[base + d] = v;
    xb[base + d] = f2bf(v);
  }
}

// ---------------------------------------------------------------------------
// LayerNorm(a + res) with gamma/beta. One wave per row; register-staged so
// yf may alias res (in-place residual stream update). Writes fp32 + bf16.
// ---------------------------------------------------------------------------
__global__ __launch_bounds__(256) void ln_kernel(
    const float* a, const float* res, const float* gamma, const float* beta,
    float* yf, unsigned short* yb) {
  int row = blockIdx.x * 8 + (threadIdx.x >> 5);
  int lane = threadIdx.x & 31;
  size_t base = (size_t)row * DMODEL;
  float t[16];
  float s = 0.0f;
  for (int j = 0; j < 16; ++j) {
    int d = j * 32 + lane;
    float v = a[base + d] + res[base + d];
    t[j] = v;
    s += v;
  }
  for (int m = 1; m < 32; m <<= 1) s += __shfl_xor(s, m, 32);
  float mu = s * (1.0f / DMODEL);
  float q = 0.0f;
  for (int j = 0; j < 16; ++j) {
    float d0 = t[j] - mu;
    q += d0 * d0;
  }
  for (int m = 1; m < 32; m <<= 1) q += __shfl_xor(q, m, 32);
  float rstd = rsqrtf(q * (1.0f / DMODEL) + 1e-5f);
  for (int j = 0; j < 16; ++j) {
    int d = j * 32 + lane;
    float v = gamma[d] * (t[j] - mu) * rstd + beta[d];
    yf[base + d] = v;
    yb[base + d] = f2bf(v);
  }
}

// ---------------------------------------------------------------------------
// V transpose: [B*Tk, D] fp32 -> [B, H, DHEAD, Tk] bf16 (key-contiguous rows
// for P*V B-fragments).
// ---------------------------------------------------------------------------
__global__ void vtrans_kernel(const float* __restrict__ vin,
                              unsigned short* __restrict__ vt, int Tk) {
  size_t i = (size_t)blockIdx.x * blockDim.x + threadIdx.x;
  size_t n = (size_t)BATCH * Tk * DMODEL;
  if (i >= n) return;
  int col = (int)(i % DMODEL);
  size_t row = i / DMODEL;
  int tt = (int)(row % Tk);
  int b = (int)(row / Tk);
  int h = col / DHEAD, d = col % DHEAD;
  vt[((size_t)(b * NHEAD + h) * DHEAD + d) * Tk + tt] = f2bf(vin[i]);
}

// ---------------------------------------------------------------------------
// GEMM: C[M,N] = A[M,K](bf16) * W[N,K]^T(bf16) + bias, optional ReLU.
// 256 threads = 8 waves; wave tile 64x64 (4x4 of 16x16 accumulators,
// 16 WMMAs per K-step of 32), WG macro-tile 256x128.
// ---------------------------------------------------------------------------
__global__ __launch_bounds__(256) void gemm_bf16_kernel(
    const unsigned short* __restrict__ A, const unsigned short* __restrict__ Bw,
    const float* __restrict__ bias, float* __restrict__ Cf,
    unsigned short* __restrict__ Cb, int M, int N, int K, int relu) {
  int lane = threadIdx.x & 31;
  int wave = threadIdx.x >> 5;
  int hs = lane >> 4;
  int ln16 = lane & 15;
  int m0 = blockIdx.y * 256 + (wave >> 1) * 64;
  int n0 = blockIdx.x * 128 + (wave & 1) * 64;

  v8f acc[4][4];
  for (int i = 0; i < 4; ++i)
    for (int j = 0; j < 4; ++j) acc[i][j] = zero8();

  const unsigned short* arow[4];
  const unsigned short* brow[4];
  for (int i = 0; i < 4; ++i) {
    arow[i] = A + (size_t)(m0 + i * 16 + ln16) * K + hs * 8;
    brow[i] = Bw + (size_t)(n0 + i * 16 + ln16) * K + hs * 8;
  }

  for (int k0 = 0; k0 < K; k0 += 32) {
    if (k0 + 32 < K) {  // near-cache prefetch of next weight K-slice
      __builtin_prefetch(brow[0] + k0 + 32, 0, 3);
      __builtin_prefetch(brow[2] + k0 + 32, 0, 3);
    }
    FragBF af[4], bf[4];
    for (int i = 0; i < 4; ++i) af[i] = ldfragA(arow[i] + k0);
    for (int j = 0; j < 4; ++j) bf[j] = ldfragB(brow[j] + k0);
    for (int i = 0; i < 4; ++i)
      for (int j = 0; j < 4; ++j)
        acc[i][j] = wmma_bf16(af[i], bf[j], acc[i][j]);
  }

  for (int i = 0; i < 4; ++i) {
    for (int j = 0; j < 4; ++j) {
      int col = n0 + j * 16 + ln16;
      float bv = bias ? bias[col] : 0.0f;
      for (int r = 0; r < 8; ++r) {
        int row = m0 + i * 16 + r + 8 * hs;
        float v = acc[i][j][r] + bv;
        if (relu) v = fmaxf(v, 0.0f);
        size_t idx = (size_t)row * N + col;
        if (Cf) Cf[idx] = v;
        if (Cb) Cb[idx] = f2bf(v);
      }
    }
  }
}

// ---------------------------------------------------------------------------
// Flash attention. One wave handles 16 query rows of one (b,h). K-tile = 32.
// Qb/Kb: [B*Tq, D] / [B*Tk, D] bf16 (head h at cols h*64..). VT: [B,H,64,Tk]
// bf16. Output written merged-heads bf16 [B*Tq, D].
// Masking matches reference: finfo.min then softmax (fully-masked -> uniform).
// ---------------------------------------------------------------------------
__global__ __launch_bounds__(256) void attn_kernel(
    const unsigned short* __restrict__ Qb, const unsigned short* __restrict__ Kb,
    const unsigned short* __restrict__ VT, const int* __restrict__ tokens,
    const unsigned char* __restrict__ encpad, unsigned short* __restrict__ Ob,
    int Tq, int Tk, int causal) {
  __shared__ unsigned short plds[8 * 16 * 32];
  const float NEGMIN = -3.402823466e38f;
  int wave = threadIdx.x >> 5;
  int lane = threadIdx.x & 31;
  int hs = lane >> 4, ln16 = lane & 15;
  int gid = blockIdx.x * 8 + wave;
  int tiles = Tq / 16;
  int bh = gid / tiles, qt = gid % tiles;
  int b = bh / NHEAD, h = bh % NHEAD;
  unsigned short* myp = plds + wave * (16 * 32);

  size_t qoff = (size_t)(b * Tq + qt * 16 + ln16) * DMODEL + h * DHEAD;
  FragBF qf0 = ldfragA(Qb + qoff + 0 + hs * 8);
  FragBF qf1 = ldfragA(Qb + qoff + 32 + hs * 8);

  float mrow[8], lrow[8];
  for (int r = 0; r < 8; ++r) { mrow[r] = NEGMIN; lrow[r] = 0.0f; }
  v8f oacc[4];
  for (int j = 0; j < 4; ++j) oacc[j] = zero8();

  int nkt = Tk / 32;
  if (causal) {
    int lim = qt / 2 + 1;
    if (lim < nkt) nkt = lim;
  }

  for (int kt = 0; kt < nkt; ++kt) {
    int kbase = kt * 32;
    v8f s0 = zero8(), s1 = zero8();
    {
      size_t kr0 = (size_t)(b * Tk + kbase + ln16) * DMODEL + h * DHEAD;
      size_t kr1 = (size_t)(b * Tk + kbase + 16 + ln16) * DMODEL + h * DHEAD;
      FragBF kf;
      kf = ldfragB(Kb + kr0 + 0 + hs * 8);  s0 = wmma_bf16(qf0, kf, s0);
      kf = ldfragB(Kb + kr0 + 32 + hs * 8); s0 = wmma_bf16(qf1, kf, s0);
      kf = ldfragB(Kb + kr1 + 0 + hs * 8);  s1 = wmma_bf16(qf0, kf, s1);
      kf = ldfragB(Kb + kr1 + 32 + hs * 8); s1 = wmma_bf16(qf1, kf, s1);
    }

    int key0 = kbase + ln16, key1 = kbase + 16 + ln16;
    bool blk0 = false, blk1 = false;
    if (tokens) {
      blk0 = (tokens[b * Tk + key0] == 0);
      blk1 = (tokens[b * Tk + key1] == 0);
    }
    if (encpad) {
      blk0 = (encpad[b * Tk + key0] != 0);
      blk1 = (encpad[b * Tk + key1] != 0);
    }

    float p0[8], p1[8];
    for (int r = 0; r < 8; ++r) {
      int qg = qt * 16 + r + 8 * hs;
      float v0 = s0[r] * 0.125f;  // 1/sqrt(64)
      float v1 = s1[r] * 0.125f;
      if (blk0 || (causal && key0 > qg)) v0 = NEGMIN;
      if (blk1 || (causal && key1 > qg)) v1 = NEGMIN;
      float rm = fmaxf(v0, v1);
      for (int m = 1; m < 16; m <<= 1) rm = fmaxf(rm, __shfl_xor(rm, m, 32));
      float mnew = fmaxf(mrow[r], rm);
      float corr = __expf(mrow[r] - mnew);
      float e0 = __expf(v0 - mnew), e1 = __expf(v1 - mnew);
      float rs = e0 + e1;
      for (int m = 1; m < 16; m <<= 1) rs += __shfl_xor(rs, m, 32);
      lrow[r] = lrow[r] * corr + rs;
      mrow[r] = mnew;
      for (int j = 0; j < 4; ++j) oacc[j][r] *= corr;
      p0[r] = e0;
      p1[r] = e1;
    }

    // Re-swizzle P (D layout) -> A fragment via LDS (same-wave, in-order DS)
    for (int r = 0; r < 8; ++r) {
      int rl = r + 8 * hs;
      myp[rl * 32 + ln16] = f2bf(p0[r]);
      myp[rl * 32 + 16 + ln16] = f2bf(p1[r]);
    }
    __builtin_amdgcn_wave_barrier();
    FragBF pf = ldfragA(myp + ln16 * 32 + hs * 8);
    __builtin_amdgcn_wave_barrier();

    for (int j = 0; j < 4; ++j) {
      size_t vb = ((size_t)(b * NHEAD + h) * DHEAD + j * 16 + ln16) * (size_t)Tk +
                  kbase + hs * 8;
      FragBF vf = ldfragB(VT + vb);
      oacc[j] = wmma_bf16(pf, vf, oacc[j]);
    }
  }

  for (int j = 0; j < 4; ++j) {
    for (int r = 0; r < 8; ++r) {
      float o = oacc[j][r] / lrow[r];
      size_t idx = (size_t)(b * Tq + qt * 16 + r + 8 * hs) * DMODEL +
                   h * DHEAD + j * 16 + ln16;
      Ob[idx] = f2bf(o);
    }
  }
}

// ---------------------------------------------------------------------------
// Host orchestration
// ---------------------------------------------------------------------------
extern "C" void kernel_launch(void* const* d_in, const int* in_sizes, int n_in,
                              void* d_out, int out_size, void* d_ws, size_t ws_size,
                              hipStream_t stream) {
  (void)in_sizes; (void)n_in; (void)out_size; (void)ws_size;

  // Inputs (setup_inputs order)
  const int*   tokens = (const int*)d_in[0];
  const float* enc    = (const float*)d_in[1];
  const unsigned char* encpad = (const unsigned char*)d_in[2];
  const float* emb    = (const float*)d_in[3];
  const float* Wq = (const float*)d_in[4];  const float* bq = (const float*)d_in[5];
  const float* Wk = (const float*)d_in[6];  const float* bk = (const float*)d_in[7];
  const float* Wv = (const float*)d_in[8];  const float* bv = (const float*)d_in[9];
  const float* Wo1 = (const float*)d_in[10]; const float* bo1 = (const float*)d_in[11];
  const float* cWq = (const float*)d_in[12]; const float* cbq = (const float*)d_in[13];
  const float* eWk = (const float*)d_in[14]; const float* ebk = (const float*)d_in[15];
  const float* eWv = (const float*)d_in[16]; const float* ebv = (const float*)d_in[17];
  const float* Wo2 = (const float*)d_in[18]; const float* bo2 = (const float*)d_in[19];
  const float* W1 = (const float*)d_in[20]; const float* b1 = (const float*)d_in[21];
  const float* W2 = (const float*)d_in[22]; const float* b2 = (const float*)d_in[23];
  const float* g1 = (const float*)d_in[24]; const float* be1 = (const float*)d_in[25];
  const float* g2 = (const float*)d_in[26]; const float* be2 = (const float*)d_in[27];
  const float* g3 = (const float*)d_in[28]; const float* be3 = (const float*)d_in[29];
  const float* Wout = (const float*)d_in[30]; const float* bout = (const float*)d_in[31];
  float* out = (float*)d_out;

  // Workspace carve-out
  size_t off = 0;
  auto alloc = [&](size_t bytes) -> void* {
    void* p = (char*)d_ws + off;
    off += (bytes + 255) & ~(size_t)255;
    return p;
  };
  const size_t NE = (size_t)NROWS * DMODEL;  // 2M elements
  float* F1 = (float*)alloc(NE * 4);         // residual stream fp32
  float* F2 = (float*)alloc(NE * 4);         // gemm fp32 scratch
  float* F3 = (float*)alloc(NE * 4);         // v fp32 scratch
  unsigned short* Xb  = (unsigned short*)alloc(NE * 2);  // residual stream bf16
  unsigned short* Qb  = (unsigned short*)alloc(NE * 2);
  unsigned short* KbB = (unsigned short*)alloc(NE * 2);
  unsigned short* VT  = (unsigned short*)alloc(NE * 2);
  unsigned short* AOb = (unsigned short*)alloc(NE * 2);
  unsigned short* ENCb = (unsigned short*)alloc(NE * 2);
  unsigned short* H1b = (unsigned short*)alloc((size_t)NROWS * 2 * DMODEL * 2);
  const size_t WSZ = (size_t)DMODEL * DMODEL;  // 262144
  unsigned short* wWq  = (unsigned short*)alloc(WSZ * 2);
  unsigned short* wWk  = (unsigned short*)alloc(WSZ * 2);
  unsigned short* wWv  = (unsigned short*)alloc(WSZ * 2);
  unsigned short* wWo1 = (unsigned short*)alloc(WSZ * 2);
  unsigned short* wcWq = (unsigned short*)alloc(WSZ * 2);
  unsigned short* weWk = (unsigned short*)alloc(WSZ * 2);
  unsigned short* weWv = (unsigned short*)alloc(WSZ * 2);
  unsigned short* wWo2 = (unsigned short*)alloc(WSZ * 2);
  unsigned short* wW1  = (unsigned short*)alloc(WSZ * 2 * 2);
  unsigned short* wW2  = (unsigned short*)alloc(WSZ * 2 * 2);
  unsigned short* wWout = (unsigned short*)alloc((size_t)VOCAB * DMODEL * 2);

  auto conv = [&](const float* src, unsigned short* dst, size_t n) {
    f2bf_kernel<<<(unsigned)((n + 255) / 256), 256, 0, stream>>>(src, dst, n);
  };
  auto gemm = [&](const unsigned short* A, const unsigned short* W,
                  const float* bias, float* Cf, unsigned short* Cb,
                  int M, int N, int K, int relu) {
    dim3 g(N / 128, M / 256);
    gemm_bf16_kernel<<<g, 256, 0, stream>>>(A, W, bias, Cf, Cb, M, N, K, relu);
  };

  // Weight conversions
  conv(Wq, wWq, WSZ);   conv(Wk, wWk, WSZ);   conv(Wv, wWv, WSZ);
  conv(Wo1, wWo1, WSZ); conv(cWq, wcWq, WSZ); conv(eWk, weWk, WSZ);
  conv(eWv, weWv, WSZ); conv(Wo2, wWo2, WSZ);
  conv(W1, wW1, WSZ * 2); conv(W2, wW2, WSZ * 2);
  conv(Wout, wWout, (size_t)VOCAB * DMODEL);
  conv(enc, ENCb, NE);

  // x0 = emb[tokens]*sqrt(D) + PE  -> F1 (fp32), Xb (bf16)
  embed_kernel<<<NROWS / 8, 256, 0, stream>>>(tokens, emb, F1, Xb);

  // --- causal self-attention ---
  gemm(Xb, wWq, bq, nullptr, Qb, NROWS, DMODEL, DMODEL, 0);
  gemm(Xb, wWk, bk, nullptr, KbB, NROWS, DMODEL, DMODEL, 0);
  gemm(Xb, wWv, bv, F3, nullptr, NROWS, DMODEL, DMODEL, 0);
  vtrans_kernel<<<(unsigned)((NE + 255) / 256), 256, 0, stream>>>(F3, VT, TLEN);
  attn_kernel<<<(BATCH * NHEAD * (TLEN / 16)) / 8, 256, 0, stream>>>(
      Qb, KbB, VT, tokens, nullptr, AOb, TLEN, TLEN, 1);
  gemm(AOb, wWo1, bo1, F2, nullptr, NROWS, DMODEL, DMODEL, 0);
  ln_kernel<<<NROWS / 8, 256, 0, stream>>>(F2, F1, g1, be1, F1, Xb);  // x1

  // --- cross-attention ---
  gemm(Xb, wcWq, cbq, nullptr, Qb, NROWS, DMODEL, DMODEL, 0);
  gemm(ENCb, weWk, ebk, nullptr, KbB, NROWS, DMODEL, DMODEL, 0);
  gemm(ENCb, weWv, ebv, F3, nullptr, NROWS, DMODEL, DMODEL, 0);
  vtrans_kernel<<<(unsigned)((NE + 255) / 256), 256, 0, stream>>>(F3, VT, SLEN);
  attn_kernel<<<(BATCH * NHEAD * (TLEN / 16)) / 8, 256, 0, stream>>>(
      Qb, KbB, VT, nullptr, encpad, AOb, TLEN, SLEN, 0);
  gemm(AOb, wWo2, bo2, F2, nullptr, NROWS, DMODEL, DMODEL, 0);
  ln_kernel<<<NROWS / 8, 256, 0, stream>>>(F2, F1, g3, be3, F1, Xb);  // x2

  // --- FFN ---
  gemm(Xb, wW1, b1, nullptr, H1b, NROWS, 2 * DMODEL, DMODEL, 1);      // ReLU
  gemm(H1b, wW2, b2, F2, nullptr, NROWS, DMODEL, 2 * DMODEL, 0);
  ln_kernel<<<NROWS / 8, 256, 0, stream>>>(F2, F1, g2, be2, F1, Xb);  // x3

  // --- LM head: logits fp32 straight to d_out ---
  gemm(Xb, wWout, bout, out, nullptr, NROWS, VOCAB, DMODEL, 0);
}
